// MultiHeadSelfAttention_16836271800929
// MI455X (gfx1250) — compile-verified, matrix operand order checked by reading, not run on hardware
//
#include <hip/hip_runtime.h>
#include <hip/hip_bf16.h>

// ---------------------------------------------------------------------------
// MI455X (gfx1250) multi-head causal self-attention with RoPE.
//   x:(1,4096,1024) fp32, Wq/Wk/Wv/Wo:(1024,1024) fp32 -> out (1,4096,1024) fp32
// Pipeline (all bf16 WMMA, fp32 accumulate):
//   K0: fp32 -> bf16 convert (x, Wq, Wk, Wv, Wo)
//   K1: QKV projection GEMM + fused RoPE; Q,K head-major, V transposed
//   K2: flash attention, 4 waves / 64 q-rows per WG in kv-lockstep so the
//       WGP read-cache amortizes K/V streams 4x; online softmax in registers
//   K3: output projection GEMM -> fp32
// ---------------------------------------------------------------------------

typedef __bf16 v16bf __attribute__((ext_vector_type(16)));
typedef __bf16 v8bf  __attribute__((ext_vector_type(8)));
typedef __bf16 v4bf  __attribute__((ext_vector_type(4)));
typedef float  v8f   __attribute__((ext_vector_type(8)));

#define D_MODEL 1024
#define SEQ     4096
#define NHEAD   16
#define DK      64

// ---- fragment load: 16x32 bf16 operand (A or B^T), row-major source --------
// Lane l holds row (l&15). K striping per CDNA5 ISA: lanes 0-15 get K chunks
// [kc+0..7] and [kc+16..23]; lanes 16-31 get [kc+8..15] and [kc+24..31].
__device__ __forceinline__ v16bf load_frag(const __bf16* rowptr, int kc, int lane) {
    const int koff = kc + ((lane >> 4) << 3);
    v8bf lo = *(const v8bf*)(rowptr + koff);
    v8bf hi = *(const v8bf*)(rowptr + koff + 16);
    v16bf r;
#pragma unroll
    for (int i = 0; i < 8; ++i) { r[i] = lo[i]; r[i + 8] = hi[i]; }
    return r;
}

__device__ __forceinline__ v8f wmma_bf16(v16bf a, v16bf b, v8f c) {
    // D = A(16x32) * B(32x16) + C, fp32 accumulate
    return __builtin_amdgcn_wmma_f32_16x16x32_bf16(false, a, false, b,
                                                   (short)0, c, false, false);
}

// ---- 64x64 per-wave GEMM core: C += A(m0..+64, :) * B(n0..+64, :)^T --------
// A: MxK row-major, B: NxK row-major (i.e. computing X * W^T), K = lda = ldb.
__device__ __forceinline__ void gemm64x64(const __bf16* __restrict__ A,
                                          const __bf16* __restrict__ B,
                                          int K, int m0, int n0, int lane,
                                          v8f acc[4][4]) {
    const __bf16* ar[4];
    const __bf16* br[4];
#pragma unroll
    for (int t = 0; t < 4; ++t) {
        ar[t] = A + (size_t)(m0 + t * 16 + (lane & 15)) * K;
        br[t] = B + (size_t)(n0 + t * 16 + (lane & 15)) * K;
    }
    for (int kc = 0; kc < K; kc += 32) {
        v16bf af[4], bfm[4];
#pragma unroll
        for (int t = 0; t < 4; ++t) {
            af[t]  = load_frag(ar[t], kc, lane);
            bfm[t] = load_frag(br[t], kc, lane);
            __builtin_prefetch(ar[t] + kc + 128, 0, 0);   // global_prefetch_b8
            __builtin_prefetch(br[t] + kc + 128, 0, 0);
        }
#pragma unroll
        for (int mt = 0; mt < 4; ++mt)
#pragma unroll
            for (int nt = 0; nt < 4; ++nt)
                acc[mt][nt] = wmma_bf16(af[mt], bfm[nt], acc[mt][nt]);
    }
}

// ---- K0: fp32 -> bf16 ------------------------------------------------------
__global__ void cvt_bf16_kernel(const float* __restrict__ src,
                                __bf16* __restrict__ dst, int n4) {
    int i = blockIdx.x * blockDim.x + threadIdx.x;
    if (i < n4) {
        float4 v = ((const float4*)src)[i];
        v4bf o;
        o[0] = (__bf16)v.x; o[1] = (__bf16)v.y;
        o[2] = (__bf16)v.z; o[3] = (__bf16)v.w;
        ((v4bf*)dst)[i] = o;
    }
}

// ---- K1: QKV projection + fused RoPE --------------------------------------
// grid (4, 32, 3): z selects Wq/Wk/Wv. 8 waves/WG, WG tile 128(M) x 256(N).
__global__ __launch_bounds__(256)
void qkv_rope_kernel(const __bf16* __restrict__ xb,
                     const __bf16* __restrict__ wq,
                     const __bf16* __restrict__ wk,
                     const __bf16* __restrict__ wv,
                     __bf16* __restrict__ QH, __bf16* __restrict__ KH,
                     __bf16* __restrict__ VT) {
    const int lane = threadIdx.x & 31;
    const int w    = threadIdx.x >> 5;
    const int m0   = blockIdx.y * 128 + (w & 1) * 64;
    const int n0   = blockIdx.x * 256 + (w >> 1) * 64;
    const int mat  = blockIdx.z;
    const __bf16* W = (mat == 0) ? wq : (mat == 1) ? wk : wv;

    v8f acc[4][4];
#pragma unroll
    for (int i = 0; i < 4; ++i)
#pragma unroll
        for (int j = 0; j < 4; ++j)
#pragma unroll
            for (int e = 0; e < 8; ++e) acc[i][j][e] = 0.0f;

    gemm64x64(xb, W, D_MODEL, m0, n0, lane, acc);

    const int hf = lane >> 4;
    const int nl = lane & 15;
#pragma unroll
    for (int mt = 0; mt < 4; ++mt)
#pragma unroll
        for (int nt = 0; nt < 4; ++nt)
#pragma unroll
            for (int r = 0; r < 8; ++r) {
                float v = acc[mt][nt][r];
                const int mg = m0 + mt * 16 + r + 8 * hf;   // sequence position
                const int d  = n0 + nt * 16 + nl;           // d_model index
                const int h  = d >> 6;
                const int dk = d & 63;
                if (mat < 2) {
                    // RoPE: partner element lives in lane^1 (pairs are 16-aligned)
                    float nb = __shfl_xor(v, 1, 32);
                    const int pair = dk >> 1;
                    // theta^{-2*pair/64} = exp(-pair * 2/64 * ln(10000))
                    float rate = __expf((float)pair * (-2.0f / 64.0f) * 9.2103403719761836f);
                    float ang = (float)mg * rate;
                    float s, c;
                    __sincosf(ang, &s, &c);
                    v = (dk & 1) ? (nb * s + v * c) : (v * c - nb * s);
                    __bf16* dst = (mat == 0) ? QH : KH;
                    dst[(size_t)(h * SEQ + mg) * DK + dk] = (__bf16)v;
                } else {
                    // V stored transposed: VT[h][dk][seq] for contiguous PV B-loads
                    VT[(size_t)(h * DK + dk) * SEQ + mg] = (__bf16)v;
                }
            }
}

// ---- K2: flash attention, causal ------------------------------------------
// 4 waves per WG handle 4 consecutive 16-row q-tiles (64 q rows). All waves
// run the SAME kv loop bound (block-max diagonal) and are kept in lockstep by
// the barriers, so their identical K/V cacheline reads coalesce in the WGP
// read cache: ~4x less L2/K-V stream traffic than independent q-tile waves.
// Fully-masked blocks for the earlier tiles are numerically benign
// (p = exp(-1e30 - m) == 0, scale == 1).
__global__ __launch_bounds__(128)
void flash_attn_kernel(const __bf16* __restrict__ QH,
                       const __bf16* __restrict__ KH,
                       const __bf16* __restrict__ VT,
                       __bf16* __restrict__ AO) {
    const int lane = threadIdx.x & 31;
    const int w    = threadIdx.x >> 5;             // wave 0..3
    const int nqb  = SEQ / 64;                      // 64 q-blocks per head
    const int h    = blockIdx.x / nqb;
    const int qb   = (blockIdx.x % nqb) * 64;       // block q base
    const int q0   = qb + w * 16;                   // this wave's q-tile
    const int row  = lane & 15;
    const int hf   = lane >> 4;

    __shared__ alignas(16) __bf16 P[4][16 * 40];    // per-wave 16x32 P tile

    const __bf16* qrow = QH + (size_t)(h * SEQ + q0 + row) * DK;
    v16bf aq0 = load_frag(qrow, 0, lane);
    v16bf aq1 = load_frag(qrow, 32, lane);

    v8f acc[4];
#pragma unroll
    for (int t = 0; t < 4; ++t)
#pragma unroll
        for (int e = 0; e < 8; ++e) acc[t][e] = 0.0f;
    float mrow[8], lrow[8];
#pragma unroll
    for (int r = 0; r < 8; ++r) { mrow[r] = -1e30f; lrow[r] = 0.0f; }

    const int kb_end = qb + 48;                     // block-max diagonal
    for (int kb = 0; kb <= kb_end; kb += 32) {
        // S = Q * K^T for kv columns [kb, kb+32)
        const __bf16* krow0 = KH + (size_t)(h * SEQ + kb + row) * DK;
        const __bf16* krow1 = krow0 + 16 * DK;
        v16bf bk0l = load_frag(krow0, 0, lane);
        v16bf bk0h = load_frag(krow0, 32, lane);
        v16bf bk1l = load_frag(krow1, 0, lane);
        v16bf bk1h = load_frag(krow1, 32, lane);
        v8f s0, s1;
#pragma unroll
        for (int e = 0; e < 8; ++e) { s0[e] = 0.0f; s1[e] = 0.0f; }
        s0 = wmma_bf16(aq0, bk0l, s0);
        s0 = wmma_bf16(aq1, bk0h, s0);
        s1 = wmma_bf16(aq0, bk1l, s1);
        s1 = wmma_bf16(aq1, bk1h, s1);

        const int n0 = kb + (lane & 15);
        const int n1 = n0 + 16;
#pragma unroll
        for (int r = 0; r < 8; ++r) {
            const int mg = q0 + r + 8 * hf;
            float e0 = s0[r] * 0.125f;                 // 1/sqrt(64)
            float e1 = s1[r] * 0.125f;
            if (n0 > mg) e0 = -1e30f;                  // causal mask
            if (n1 > mg) e1 = -1e30f;
            float mx = fmaxf(e0, e1);
#pragma unroll
            for (int d = 1; d < 16; d <<= 1) mx = fmaxf(mx, __shfl_xor(mx, d, 32));
            const float mnew  = fmaxf(mrow[r], mx);
            const float scale = __expf(mrow[r] - mnew);
            mrow[r] = mnew;
            const float p0 = __expf(e0 - mnew);
            const float p1 = __expf(e1 - mnew);
            float rs = p0 + p1;
#pragma unroll
            for (int d = 1; d < 16; d <<= 1) rs += __shfl_xor(rs, d, 32);
            lrow[r] = lrow[r] * scale + rs;
#pragma unroll
            for (int t = 0; t < 4; ++t) acc[t][r] *= scale;
            const int m = r + 8 * hf;
            P[w][m * 40 + (lane & 15)]      = (__bf16)p0;  // C-layout -> (m,k)
            P[w][m * 40 + 16 + (lane & 15)] = (__bf16)p1;
        }
        __syncthreads();
        // re-read P as an A-operand fragment and do O += P * V
        v16bf pa = load_frag(&P[w][(lane & 15) * 40], 0, lane);
#pragma unroll
        for (int nt = 0; nt < 4; ++nt) {
            const __bf16* vrow = VT + (size_t)(h * DK + nt * 16 + row) * SEQ + kb;
            v16bf bv = load_frag(vrow, 0, lane);
            acc[nt] = wmma_bf16(pa, bv, acc[nt]);
        }
        __syncthreads();
    }

#pragma unroll
    for (int nt = 0; nt < 4; ++nt)
#pragma unroll
        for (int r = 0; r < 8; ++r) {
            const int mg = q0 + r + 8 * hf;
            const float o = acc[nt][r] / lrow[r];
            AO[(size_t)mg * D_MODEL + h * DK + nt * 16 + (lane & 15)] = (__bf16)o;
        }
}

// ---- K3: output projection, fp32 result -----------------------------------
__global__ __launch_bounds__(256)
void out_proj_kernel(const __bf16* __restrict__ AO,
                     const __bf16* __restrict__ wo,
                     float* __restrict__ out) {
    const int lane = threadIdx.x & 31;
    const int w    = threadIdx.x >> 5;
    const int m0   = blockIdx.y * 128 + (w & 1) * 64;
    const int n0   = blockIdx.x * 256 + (w >> 1) * 64;

    v8f acc[4][4];
#pragma unroll
    for (int i = 0; i < 4; ++i)
#pragma unroll
        for (int j = 0; j < 4; ++j)
#pragma unroll
            for (int e = 0; e < 8; ++e) acc[i][j][e] = 0.0f;

    gemm64x64(AO, wo, D_MODEL, m0, n0, lane, acc);

    const int hf = lane >> 4;
    const int nl = lane & 15;
#pragma unroll
    for (int mt = 0; mt < 4; ++mt)
#pragma unroll
        for (int nt = 0; nt < 4; ++nt)
#pragma unroll
            for (int r = 0; r < 8; ++r)
                out[(size_t)(m0 + mt * 16 + r + 8 * hf) * D_MODEL +
                    (n0 + nt * 16 + nl)] = acc[mt][nt][r];
}

// ---------------------------------------------------------------------------
extern "C" void kernel_launch(void* const* d_in, const int* in_sizes, int n_in,
                              void* d_out, int out_size, void* d_ws, size_t ws_size,
                              hipStream_t stream) {
    const float* x  = (const float*)d_in[0];
    const float* Wq = (const float*)d_in[1];
    const float* Wk = (const float*)d_in[2];
    const float* Wv = (const float*)d_in[3];
    const float* Wo = (const float*)d_in[4];

    char* p = (char*)d_ws;
    const size_t XB = (size_t)SEQ * D_MODEL * sizeof(__bf16);      // 8 MB
    const size_t WB = (size_t)D_MODEL * D_MODEL * sizeof(__bf16);  // 2 MB
    const size_t HB = (size_t)NHEAD * SEQ * DK * sizeof(__bf16);   // 8 MB
    __bf16* xb  = (__bf16*)p; p += XB;
    __bf16* wqb = (__bf16*)p; p += WB;
    __bf16* wkb = (__bf16*)p; p += WB;
    __bf16* wvb = (__bf16*)p; p += WB;
    __bf16* wob = (__bf16*)p; p += WB;
    __bf16* QH  = (__bf16*)p; p += HB;
    __bf16* KH  = (__bf16*)p; p += HB;
    __bf16* VT  = (__bf16*)p; p += HB;
    __bf16* AO  = (__bf16*)p; p += XB;

    auto cvt = [&](const float* s, __bf16* d, int n) {
        int n4 = n / 4;
        cvt_bf16_kernel<<<(n4 + 255) / 256, 256, 0, stream>>>(s, d, n4);
    };
    cvt(x,  xb,  SEQ * D_MODEL);
    cvt(Wq, wqb, D_MODEL * D_MODEL);
    cvt(Wk, wkb, D_MODEL * D_MODEL);
    cvt(Wv, wvb, D_MODEL * D_MODEL);
    cvt(Wo, wob, D_MODEL * D_MODEL);

    qkv_rope_kernel<<<dim3(D_MODEL / 256, SEQ / 128, 3), 256, 0, stream>>>(
        xb, wqb, wkb, wvb, QH, KH, VT);

    flash_attn_kernel<<<dim3(NHEAD * (SEQ / 64)), 128, 0, stream>>>(QH, KH, VT, AO);

    out_proj_kernel<<<dim3(D_MODEL / 256, SEQ / 128, 1), 256, 0, stream>>>(
        AO, wob, (float*)d_out);
}
